// MultiHeadAttention_45775761441132
// MI455X (gfx1250) — matrix operand order repaired and checked by reading, not compiled, not run on hardware
//
#include <hip/hip_runtime.h>
#include <cstdint>
#include <cstddef>

// ---------------------------------------------------------------------------
// CDNA5 (gfx1250) wave32 WMMA multi-head attention.
//   B=2, S=2048, E=1024, H=16, D=64
// Pipeline:
//   0) cvt_kernel x3    : one-shot fp32 -> bf16 of x, w_kernel, fc_kernel
//                         (weights reused 256x / 48x -> never convert in hot loop)
//   1) qkv_wmma_kernel  : Xbf @ Wbf + w_bias -> Q (scaled, [B,H,S,D] bf16),
//                         K^T ([B,H,D,S] bf16), V ([B,H,S,D] bf16)
//   2) flash_attn_kernel: streaming softmax(QK^T)V per 16-row q tile (WMMA)
//   3) proj_wmma_kernel : attn(bf16) @ FCbf + fc_bias -> f32 output
// ---------------------------------------------------------------------------

typedef __attribute__((ext_vector_type(16))) __bf16        v16bf;
typedef __attribute__((ext_vector_type(8)))  float         v8f;
typedef __attribute__((ext_vector_type(4)))  unsigned int  u32x4;
typedef __attribute__((ext_vector_type(4)))  float         f32x4;

// A/B fragment for v_wmma_f32_16x16x32_bf16: 16 bf16 per lane = 8 VGPRs.
union Frag {
  v16bf v;
  unsigned short u[16];
  u32x4 q[2];
};

#define WMMA_BF16(A, Bm, C) \
  __builtin_amdgcn_wmma_f32_16x16x32_bf16(false, (A), false, (Bm), (short)0, (C), false, false)

// Native RNE convert (hardware bf16 cvt on gfx1250).
__device__ __forceinline__ unsigned short f2bf(float f) {
  union { __bf16 h; unsigned short u; } cv;
  cv.h = (__bf16)f;
  return cv.u;
}

// ---------------------------------------------------------------------------
// Kernel 0: bulk fp32 -> bf16 (8 elements/thread). n8 = n/8.
// ---------------------------------------------------------------------------
__global__ __launch_bounds__(256) void cvt_kernel(const float* __restrict__ src,
                                                  unsigned short* __restrict__ dst,
                                                  int n8) {
  const int i = blockIdx.x * 256 + threadIdx.x;
  if (i >= n8) return;
  const f32x4 a = ((const f32x4*)src)[2 * i];
  const f32x4 b = ((const f32x4*)src)[2 * i + 1];
  Frag t;
#pragma unroll
  for (int j = 0; j < 4; ++j) { t.u[j] = f2bf(a[j]); t.u[4 + j] = f2bf(b[j]); }
  ((u32x4*)dst)[i] = t.q[0];
}

// ---------------------------------------------------------------------------
// Kernel 1: QKV projection. One wave -> 16(M) x 64(N) strip; since D=64 each
// strip is exactly one (head, Q|K|V-segment). 256 row tiles x 48 strips
// -> 12288 waves -> 1536 blocks of 8 waves. Hot loop: pure bf16 loads + WMMA.
// ---------------------------------------------------------------------------
__global__ __launch_bounds__(256) void qkv_wmma_kernel(
    const unsigned short* __restrict__ X,   // [4096,1024] bf16
    const unsigned short* __restrict__ W,   // [1024,3072] bf16
    const float* __restrict__ bias,         // [3072]
    unsigned short* __restrict__ Q,         // [B*H,2048,64] bf16 (pre-scaled 1/8)
    unsigned short* __restrict__ KT,        // [B*H,64,2048] bf16 (transposed)
    unsigned short* __restrict__ V) {       // [B*H,2048,64] bf16
  const int lane = threadIdx.x & 31;
  const int lo = lane & 15, hi = lane >> 4;
  const int wave = blockIdx.x * 8 + (threadIdx.x >> 5);
  const int tm = wave / 48, tn = wave % 48;
  const int m0 = tm * 16, n0 = tn * 64;

  v8f acc[4] = {};
  const unsigned short* xrow = X + (size_t)(m0 + lo) * 1024;

  for (int k = 0; k < 1024; k += 32) {
    // A: 16x32, lane lo = row m; K = hi*8+{0..7} and 16+hi*8+{0..7}
    Frag a;
    a.q[0] = *(const u32x4*)(xrow + k + hi * 8);
    a.q[1] = *(const u32x4*)(xrow + k + 16 + hi * 8);
    // B: 32x16 per tile, lane col n=lo; K = hi*8+{0..7} and 16+hi*8+{0..7}
    const unsigned short* wp = W + (size_t)(k + 8 * hi) * 3072 + n0 + lo;
    if (k + 32 < 1024) {  // speculative prefetch of next iteration's panels
      __builtin_prefetch(wp + (size_t)32 * 3072, 0, 0);
      __builtin_prefetch(xrow + k + 32, 0, 0);
    }
#pragma unroll
    for (int t = 0; t < 4; ++t) {
      Frag bf;
      for (int s = 0; s < 16; ++s) bf.u[s] = wp[(size_t)((s & 7) + 16 * (s >> 3)) * 3072 + t * 16];
      acc[t] = WMMA_BF16(a.v, bf.v, acc[t]);
    }
  }

  // Epilogue: C layout -> lane lo = col, VGPR g = row (lanes>=16: row+8)
  const int seg = tn / 16;  // 0:Q 1:K 2:V
  const int h   = tn % 16;  // head
#pragma unroll
  for (int t = 0; t < 4; ++t) {
    const int d = t * 16 + lo;
    const float bval = bias[n0 + t * 16 + lo];
#pragma unroll
    for (int g = 0; g < 8; ++g) {
      const float val = acc[t][g] + bval;
      const int r = m0 + g + 8 * hi;
      const int b = r >> 11, s = r & 2047;
      const size_t bh = (size_t)(b * 16 + h);
      if (seg == 0)      Q [(bh * 2048 + s) * 64 + d] = f2bf(val * 0.125f); // fold 1/sqrt(D)
      else if (seg == 1) KT[(bh * 64 + d) * 2048 + s] = f2bf(val);
      else               V [(bh * 2048 + s) * 64 + d] = f2bf(val);
    }
  }
}

// ---------------------------------------------------------------------------
// Kernel 2: streaming attention. One wave per (b,h,q-tile of 16). 32 keys per
// iteration: 4 QK^T WMMAs, mask quirk (masked -> 1e-9, still softmaxed),
// online softmax, P re-layout via per-wave LDS tile, 4 PV WMMAs.
// 32 bh * 128 qtiles = 4096 waves -> 1024 blocks of 4 waves.
// ---------------------------------------------------------------------------
__global__ __launch_bounds__(128) void flash_attn_kernel(
    const unsigned short* __restrict__ Q,
    const unsigned short* __restrict__ KT,
    const unsigned short* __restrict__ V,
    const unsigned char*  __restrict__ mask,  // [B,2048,2048] bool (1 = masked)
    unsigned short* __restrict__ O) {         // [B,2048,1024] bf16 (head-merged)
  __shared__ unsigned short plds[4][16 * 32];  // per-wave P staging tile

  const int lane  = threadIdx.x & 31;
  const int lo = lane & 15, hi = lane >> 4;
  const int wslot = threadIdx.x >> 5;
  const int wave  = blockIdx.x * 4 + wslot;
  const int bh = wave >> 7;
  const int q0 = (wave & 127) * 16;
  const int b = bh >> 4, h = bh & 15;

  const unsigned short* Qb   = Q  + (size_t)bh * 2048 * 64;
  const unsigned short* KTb  = KT + (size_t)bh * 64 * 2048;
  const unsigned short* Vb   = V  + (size_t)bh * 2048 * 64;
  const unsigned char*  mrow = mask + (size_t)b * 2048 * 2048 + (size_t)(q0 + 8 * hi) * 2048;

  // Q A-fragments, d-chunks 0..31 and 32..63 (already pre-scaled by 1/8)
  Frag qa[2];
  {
    const unsigned short* qp = Qb + (size_t)(q0 + lo) * 64 + hi * 8;
#pragma unroll
    for (int c = 0; c < 2; ++c) {
      qa[c].q[0] = *(const u32x4*)(qp + 32 * c);
      qa[c].q[1] = *(const u32x4*)(qp + 32 * c + 16);
    }
  }

  float m_run[8], l_run[8];
#pragma unroll
  for (int g = 0; g < 8; ++g) { m_run[g] = -3.0e38f; l_run[g] = 0.0f; }
  v8f o[4] = {};

  for (int kc = 0; kc < 2048; kc += 32) {
    // Prefetch next key chunk (KT rows and V rows, L2-resident streams).
    if (kc + 32 < 2048) {
      __builtin_prefetch(KTb + (size_t)(lo + 16 * hi) * 2048 + kc + 32, 0, 0);
      __builtin_prefetch(Vb + (size_t)(kc + 32 + lo + 16 * hi) * 64, 0, 0);
    }

    // ---- scores: two 16x16 key sub-tiles, accumulate over D=64 (2 chunks)
    v8f sc[2];
#pragma unroll
    for (int st = 0; st < 2; ++st) {
      v8f s_acc = {};
#pragma unroll
      for (int c = 0; c < 2; ++c) {
        Frag kb;  // B: lane holds key kc+16st+lo, d = 32c+hi*8+{0..7}, 32c+16+hi*8+{0..7}
        const unsigned short* kp =
            KTb + (size_t)(32 * c + 8 * hi) * 2048 + kc + 16 * st + lo;
        for (int s = 0; s < 16; ++s) kb.u[s] = kp[(size_t)((s & 7) + 16 * (s >> 3)) * 2048];
        s_acc = WMMA_BF16(qa[c].v, kb.v, s_acc);
      }
      sc[st] = s_acc;
    }

    // ---- mask quirk + online softmax (rows split across lane halves)
    float alpha[8];
#pragma unroll
    for (int g = 0; g < 8; ++g) {
      float v0 = sc[0][g], v1 = sc[1][g];
      const unsigned char* mp = mrow + (size_t)g * 2048 + kc + lo;
      if (mp[0])  v0 = 1e-9f;   // masked scores are REPLACED, still softmaxed
      if (mp[16]) v1 = 1e-9f;
      float mx = fmaxf(v0, v1);
#pragma unroll
      for (int off = 1; off < 16; off <<= 1) mx = fmaxf(mx, __shfl_xor(mx, off, 16));
      const float mnew = fmaxf(m_run[g], mx);
      const float a    = __expf(m_run[g] - mnew);   // 0 on first iter (underflow)
      const float p0   = __expf(v0 - mnew);
      const float p1   = __expf(v1 - mnew);
      float sum = p0 + p1;
#pragma unroll
      for (int off = 1; off < 16; off <<= 1) sum += __shfl_xor(sum, off, 16);
      l_run[g] = l_run[g] * a + sum;
      m_run[g] = mnew;
      alpha[g] = a;
      // C-layout -> LDS (m, keyloc) for A-layout reload
      plds[wslot][(g + 8 * hi) * 32 + lo]      = f2bf(p0);
      plds[wslot][(g + 8 * hi) * 32 + 16 + lo] = f2bf(p1);
    }

    // ---- P A-fragment from LDS (DS ops in-order within wave)
    Frag pf;
    const unsigned short* lp = &plds[wslot][lo * 32 + hi * 8];
    pf.q[0] = *(const u32x4*)(lp);
    pf.q[1] = *(const u32x4*)(lp + 16);

    // ---- rescale running output, then P @ V
#pragma unroll
    for (int t = 0; t < 4; ++t)
#pragma unroll
      for (int g = 0; g < 8; ++g) o[t][g] *= alpha[g];
#pragma unroll
    for (int t = 0; t < 4; ++t) {
      Frag vf;  // B: lane holds d = t*16+lo, keys kc+hi*8+{0..7}, kc+16+hi*8+{0..7}
      const unsigned short* vp = Vb + (size_t)(kc + 8 * hi) * 64 + t * 16 + lo;
      for (int s = 0; s < 16; ++s) vf.u[s] = vp[((s & 7) + 16 * (s >> 3)) * 64];
      o[t] = WMMA_BF16(pf.v, vf.v, o[t]);
    }
  }

  // ---- finalize: divide by l, write [B,S,E] with heads merged back
#pragma unroll
  for (int t = 0; t < 4; ++t)
#pragma unroll
    for (int g = 0; g < 8; ++g) {
      const float val = o[t][g] / l_run[g];
      const int r = q0 + g + 8 * hi;
      O[(size_t)(b * 2048 + r) * 1024 + h * 64 + t * 16 + lo] = f2bf(val);
    }
}

// ---------------------------------------------------------------------------
// Kernel 3: output projection. Both operands already bf16. 256*16 = 4096 waves.
// ---------------------------------------------------------------------------
__global__ __launch_bounds__(256) void proj_wmma_kernel(
    const unsigned short* __restrict__ A,   // [4096,1024] bf16 (attn out)
    const unsigned short* __restrict__ W,   // [1024,1024] bf16
    const float* __restrict__ bias,         // [1024]
    float* __restrict__ out) {              // [4096,1024] f32
  const int lane = threadIdx.x & 31;
  const int lo = lane & 15, hi = lane >> 4;
  const int wave = blockIdx.x * 8 + (threadIdx.x >> 5);
  const int tm = wave / 16, tn = wave % 16;
  const int m0 = tm * 16, n0 = tn * 64;

  v8f acc[4] = {};
  const unsigned short* arow = A + (size_t)(m0 + lo) * 1024;

  for (int k = 0; k < 1024; k += 32) {
    Frag a;
    const unsigned short* ap = arow + k + hi * 8;
    a.q[0] = *(const u32x4*)(ap);
    a.q[1] = *(const u32x4*)(ap + 16);
    const unsigned short* wp = W + (size_t)(k + 8 * hi) * 1024 + n0 + lo;
    if (k + 32 < 1024) {
      __builtin_prefetch(wp + (size_t)32 * 1024, 0, 0);
      __builtin_prefetch(ap + 32, 0, 0);
    }
#pragma unroll
    for (int t = 0; t < 4; ++t) {
      Frag bf;
      for (int s = 0; s < 16; ++s) bf.u[s] = wp[(size_t)((s & 7) + 16 * (s >> 3)) * 1024 + t * 16];
      acc[t] = WMMA_BF16(a.v, bf.v, acc[t]);
    }
  }

#pragma unroll
  for (int t = 0; t < 4; ++t) {
    const float bval = bias[n0 + t * 16 + lo];
#pragma unroll
    for (int g = 0; g < 8; ++g)
      out[(size_t)(m0 + g + 8 * hi) * 1024 + n0 + t * 16 + lo] = acc[t][g] + bval;
  }
}

// ---------------------------------------------------------------------------
extern "C" void kernel_launch(void* const* d_in, const int* in_sizes, int n_in,
                              void* d_out, int out_size, void* d_ws, size_t ws_size,
                              hipStream_t stream) {
  (void)in_sizes; (void)n_in; (void)out_size; (void)ws_size;
  const float*         x    = (const float*)d_in[0];
  const unsigned char* mask = (const unsigned char*)d_in[1];  // jax bool_: 1 byte
  const float*         wk   = (const float*)d_in[2];
  const float*         wb   = (const float*)d_in[3];
  const float*         fck  = (const float*)d_in[4];
  const float*         fcb  = (const float*)d_in[5];

  // Workspace (bf16 elements):
  //   Q/KT/V/AT : 4 x 4,194,304   (B*H*S*D)
  //   Xbf       : 4,194,304       (B*S*E)
  //   Wbf       : 3,145,728       (E*3E)
  //   FCbf      : 1,048,576       (E*E)
  // total 25,165,824 ushort = 48 MB
  unsigned short* Q    = (unsigned short*)d_ws;
  unsigned short* KT   = Q    + (size_t)4194304;
  unsigned short* V    = KT   + (size_t)4194304;
  unsigned short* AT   = V    + (size_t)4194304;
  unsigned short* Xbf  = AT   + (size_t)4194304;
  unsigned short* Wbf  = Xbf  + (size_t)4194304;
  unsigned short* FCbf = Wbf  + (size_t)3145728;

  // One-shot fp32 -> bf16 of reused operands (repaid 48x-256x in GEMM loops).
  cvt_kernel<<<2048, 256, 0, stream>>>(x,   Xbf,  4194304 / 8);
  cvt_kernel<<<1536, 256, 0, stream>>>(wk,  Wbf,  3145728 / 8);
  cvt_kernel<<< 512, 256, 0, stream>>>(fck, FCbf, 1048576 / 8);

  qkv_wmma_kernel  <<<1536, 256, 0, stream>>>(Xbf, Wbf, wb, Q, KT, V);
  flash_attn_kernel<<<1024, 128, 0, stream>>>(Q, KT, V, mask, AT);
  proj_wmma_kernel <<< 512, 256, 0, stream>>>(AT, FCbf, fcb, (float*)d_out);
}
